// ConvAttnPool_10273561772581
// MI455X (gfx1250) — matrix operand order repaired and checked by reading, not compiled, not run on hardware
//
#include <hip/hip_runtime.h>
#include <hip/hip_bf16.h>
#include <float.h>

typedef __attribute__((ext_vector_type(16))) _Float16 v16h;
typedef __attribute__((ext_vector_type(8)))  float    v8f;

constexpr int Bb = 8, Ll = 2500, Dd = 100, Ff = 50, Yy = 8000, Kk = 9, PADc = 4;
constexpr int LP = 2528;           // padded length (79*32, >= 160*16 is NOT needed; reads capped at 2528)
constexpr int FP = 64;             // padded feature dim
constexpr int NLT16 = 157;         // 16-wide l tiles covering 2500
constexpr int NLCH = 79;           // conv chunks of 32
constexpr int WVS = 8;             // waves per block in fused kernel
constexpr int TPW = 20;            // l-tiles per wave (8*20 = 160 >= 157)
constexpr long long H16_ELEMS = (long long)Bb * LP * FP;

// ---------------- kernel 0: zero the padded h buffers ----------------
__global__ void k0_zero(unsigned* p, int n) {
  int i = blockIdx.x * blockDim.x + threadIdx.x;
  if (i < n) p[i] = 0u;
}

// ---------------- kernel 1: embed + conv1d(same) + tanh --> h16, hT16 ----------------
__global__ void k1_conv(const int* __restrict__ x, const float* __restrict__ embed_W,
                        const float* __restrict__ conv_w, const float* __restrict__ conv_b,
                        _Float16* __restrict__ h16, _Float16* __restrict__ hT16) {
  __shared__ float e_s[40][Dd];            // 32 outputs + K-1 halo rows
  int blk = blockIdx.x;
  int b = blk / NLCH;
  int l0 = (blk % NLCH) * 32;
  int tid = threadIdx.x;

  for (int idx = tid; idx < 40 * Dd; idx += blockDim.x) {
    int row = idx / Dd, d = idx % Dd;
    int l = l0 + row - PADc;
    float v = 0.f;
    if (l >= 0 && l < Ll) {
      int tok = x[b * Ll + l];
      v = embed_W[(size_t)tok * Dd + d];
    }
    e_s[row][d] = v;
  }
  __syncthreads();

  for (int idx = tid; idx < 32 * Ff; idx += blockDim.x) {
    int li = idx / Ff, f = idx % Ff;
    int l = l0 + li;
    if (l >= Ll) continue;
    float acc = conv_b[f];
    for (int k = 0; k < Kk; ++k) {
      const float* er = &e_s[li + k][0];
      const float* wr = conv_w + (size_t)f * (Dd * Kk) + k;
      #pragma unroll 4
      for (int d = 0; d < Dd; ++d) acc = fmaf(er[d], wr[d * Kk], acc);
    }
    _Float16 hh = (_Float16)tanhf(acc);
    h16[((size_t)b * LP + l) * FP + f] = hh;   // row-major (L, 64)
    hT16[((size_t)b * FP + f) * LP + l] = hh;  // transposed (64, L)
  }
}

// ---------------- fused kernel: scores (WMMA, register-resident) -> flash softmax
// ---------------- -> single alpha write -> m = alpha @ h (WMMA) -> yhat ----------------
__global__ void __launch_bounds__(256, 1)
kF_attn(const float* __restrict__ U_w, const _Float16* __restrict__ h16,
        const _Float16* __restrict__ hT16, const float* __restrict__ final_w,
        const float* __restrict__ final_b, float* __restrict__ alphaG,
        float* __restrict__ yhat) {
  __shared__ float smax[WVS][16], ssum[WVS][16];
  __shared__ float lmax[16], lsum[16];
  __shared__ _Float16 stage[WVS][16][34];      // 34 = 32 + pad (odd dword stride)
  __shared__ float mred[WVS][4][8][32];        // [wave][ftile][vgpr][lane]
  __shared__ float mfin[16][FP];

  int b  = blockIdx.x / (Yy / 16);
  int yt = blockIdx.x % (Yy / 16);
  int tid = threadIdx.x;
  int wave = tid >> 5, lane = tid & 31;
  int grp = lane >> 4, ln16 = lane & 15;
  int y = yt * 16 + ln16;

  // A operand for GEMM1: U tile, contiguous k order (f = grp*16 + j)
  v16h a0, a1;
  #pragma unroll
  for (int j = 0; j < 16; ++j) {
    int f = j + grp * 16;                             // 0..31
    a0[j] = (_Float16)U_w[(size_t)y * Ff + f];
    int f1 = f + 32;                                  // 32..63 (>=50 zero padded)
    a1[j] = (f1 < Ff) ? (_Float16)U_w[(size_t)y * Ff + f1] : (_Float16)0.f;
  }

  const v8f z = {0.f,0.f,0.f,0.f,0.f,0.f,0.f,0.f};

  // ---- phase 1: scores into registers + online (max,sum) ----
  v8f sc[TPW];
  float rm[8], rs[8];
  #pragma unroll
  for (int r = 0; r < 8; ++r) { rm[r] = -FLT_MAX; rs[r] = 0.f; }

  #pragma unroll
  for (int i = 0; i < TPW; ++i) {
    int lt = wave * TPW + i;                 // wave-uniform
    v8f c = z;
    if (lt < NLT16) {                        // wave-uniform branch: EXEC stays all-1
      int col = lt * 16 + ln16;
      const v16h* bp = reinterpret_cast<const v16h*>(
          h16 + ((size_t)b * LP + col) * FP + grp * 16);
      v16h b0 = bp[0];                       // f = grp*16 ..
      v16h b1 = bp[2];                       // f = 32+grp*16 ..
      c = __builtin_amdgcn_wmma_f32_16x16x32_f16(false, a0, false, b0, (short)0, c, false, false);
      c = __builtin_amdgcn_wmma_f32_16x16x32_f16(false, a1, false, b1, (short)0, c, false, false);
      if (col < Ll) {
        #pragma unroll
        for (int r = 0; r < 8; ++r) {
          float s = c[r];
          float mo = rm[r];
          float mn = fmaxf(mo, s);
          rs[r] = rs[r] * __expf(mo - mn) + __expf(s - mn);
          rm[r] = mn;
        }
      }
    }
    sc[i] = c;
  }

  // reduce (max,sum) across 16 lanes of each half-wave
  #pragma unroll
  for (int off = 1; off < 16; off <<= 1) {
    #pragma unroll
    for (int r = 0; r < 8; ++r) {
      float om = __shfl_xor(rm[r], off, 32);
      float os = __shfl_xor(rs[r], off, 32);
      float mn = fmaxf(rm[r], om);
      rs[r] = rs[r] * __expf(rm[r] - mn) + os * __expf(om - mn);
      rm[r] = mn;
    }
  }
  if (ln16 == 0) {
    #pragma unroll
    for (int r = 0; r < 8; ++r) {
      smax[wave][grp * 8 + r] = rm[r];
      ssum[wave][grp * 8 + r] = rs[r];
    }
  }
  __syncthreads();
  if (tid < 16) {
    float m = -FLT_MAX, s = 0.f;
    for (int w = 0; w < WVS; ++w) {
      float mw = smax[w][tid], sw = ssum[w][tid];
      float mn = fmaxf(m, mw);
      s = s * __expf(m - mn) + sw * __expf(mw - mn);
      m = mn;
    }
    lmax[tid] = m;
    lsum[tid] = s;
  }
  __syncthreads();

  // ---- phase 2: normalize from registers, write alpha once, m GEMM ----
  v8f macc[4] = {z, z, z, z};
  size_t arow0 = ((size_t)b * Yy + yt * 16) * Ll;
  #pragma unroll
  for (int p = 0; p < TPW / 2; ++p) {
    int lt0 = wave * TPW + 2 * p;            // wave-uniform
    if (lt0 >= NLT16) continue;
    // stage normalized alpha (f16) for tiles lt0, lt0+1; write f32 alpha to global
    #pragma unroll
    for (int t = 0; t < 2; ++t) {
      int i = 2 * p + t;
      int lt = lt0 + t;
      int col = lt * 16 + ln16;
      bool valid = (lt < NLT16) && (col < Ll);
      #pragma unroll
      for (int r = 0; r < 8; ++r) {
        int row = grp * 8 + r;
        float al = 0.f;
        if (valid) {
          al = __expf(sc[i][r] - lmax[row]) / lsum[row];
          alphaG[arow0 + (size_t)row * Ll + col] = al;
        }
        stage[wave][row][t * 16 + ln16] = (_Float16)al;
      }
    }
    // A operand from wave-private stage (DS ops in-order within a wave)
    v16h a;
    #pragma unroll
    for (int j = 0; j < 16; ++j) {
      int koff = j + grp * 16;
      a[j] = stage[wave][ln16][koff];
    }
    int lbase = lt0 * 16 + grp * 16;         // K chunk base (max 2496+16+16 = 2528 = LP)
    __builtin_prefetch(hT16 + ((size_t)b * FP + ln16) * LP + lbase + 32, 0, 1);
    #pragma unroll
    for (int ft = 0; ft < 4; ++ft) {
      int f = ft * 16 + ln16;
      v16h bv = *reinterpret_cast<const v16h*>(
          hT16 + ((size_t)b * FP + f) * LP + lbase);
      macc[ft] = __builtin_amdgcn_wmma_f32_16x16x32_f16(false, a, false, bv, (short)0,
                                                        macc[ft], false, false);
    }
  }

  #pragma unroll
  for (int ft = 0; ft < 4; ++ft) {
    #pragma unroll
    for (int r = 0; r < 8; ++r) mred[wave][ft][r][lane] = macc[ft][r];
  }
  __syncthreads();
  for (int idx = tid; idx < 4 * 8 * 32; idx += blockDim.x) {
    int ft = idx >> 8, r = (idx >> 5) & 7, ln = idx & 31;
    float s = 0.f;
    #pragma unroll
    for (int w = 0; w < WVS; ++w) s += mred[w][ft][r][ln];
    mfin[r + ((ln >= 16) ? 8 : 0)][ft * 16 + (ln & 15)] = s;
  }
  __syncthreads();
  if (tid < 16) {
    int yo = yt * 16 + tid;
    float acc = final_b[yo];
    for (int f = 0; f < Ff; ++f) acc = fmaf(final_w[(size_t)yo * Ff + f], mfin[tid][f], acc);
    yhat[(size_t)b * Yy + yo] = acc;
  }
}

// ---------------- kernel 4: mean BCE-with-logits ----------------
__global__ void k4_loss(const float* __restrict__ yhat, const float* __restrict__ target,
                        float* __restrict__ loss) {
  __shared__ float part[256];
  int tid = threadIdx.x;
  float acc = 0.f;
  for (int i = tid; i < Bb * Yy; i += blockDim.x) {
    float yh = yhat[i], t = target[i];
    acc += fmaxf(yh, 0.f) - yh * t + log1pf(__expf(-fabsf(yh)));
  }
  part[tid] = acc;
  __syncthreads();
  for (int s = 128; s > 0; s >>= 1) {
    if (tid < s) part[tid] += part[tid + s];
    __syncthreads();
  }
  if (tid == 0) loss[0] = part[0] / (float)(Bb * Yy);
}

extern "C" void kernel_launch(void* const* d_in, const int* in_sizes, int n_in,
                              void* d_out, int out_size, void* d_ws, size_t ws_size,
                              hipStream_t stream) {
  const int*   x        = (const int*)d_in[0];
  const float* target   = (const float*)d_in[1];
  const float* embed_W  = (const float*)d_in[2];
  const float* conv_w   = (const float*)d_in[3];
  const float* conv_b   = (const float*)d_in[4];
  const float* U_w      = (const float*)d_in[5];
  const float* final_w  = (const float*)d_in[6];
  const float* final_b  = (const float*)d_in[7];

  float* out    = (float*)d_out;
  float* yhat   = out;                            // (B, Y) flat
  float* loss   = out + (size_t)Bb * Yy;
  float* alphaG = out + (size_t)Bb * Yy + 1;      // (B, Y, L) flat

  _Float16* h16  = (_Float16*)d_ws;
  _Float16* hT16 = h16 + H16_ELEMS;

  int zn = (int)(2 * H16_ELEMS * sizeof(_Float16) / 4);   // u32 count to zero
  hipLaunchKernelGGL(k0_zero, dim3((zn + 255) / 256), dim3(256), 0, stream,
                     (unsigned*)d_ws, zn);
  hipLaunchKernelGGL(k1_conv, dim3(Bb * NLCH), dim3(256), 0, stream,
                     x, embed_W, conv_w, conv_b, h16, hT16);
  hipLaunchKernelGGL(kF_attn, dim3(Bb * (Yy / 16)), dim3(256), 0, stream,
                     U_w, h16, hT16, final_w, final_b, alphaG, yhat);
  hipLaunchKernelGGL(k4_loss, dim3(1), dim3(256), 0, stream, yhat, target, loss);
}
